// Block_87737591923412
// MI455X (gfx1250) — compile-verified, matrix operand order checked by reading, not run on hardware
//
#include <hip/hip_runtime.h>
#include <math.h>

// ---------------------------------------------------------------------------
// MI455X (gfx1250) implementation: all matmuls via v_wmma_f32_16x16x32_bf16,
// fused flash attention (scores never touch HBM), async-to-LDS staging,
// global prefetch on weight panels.
// ---------------------------------------------------------------------------

typedef __attribute__((ext_vector_type(16))) __bf16 v16bf;
typedef __attribute__((ext_vector_type(8)))  float  v8f;
typedef __attribute__((ext_vector_type(4)))  unsigned int u32x4;

#define C_DIM   512
#define N_TOK   4096
#define B_SZ    2
#define HGT     64
#define WID     64
#define M_TOK   1024
#define HEADS   8
#define HD      64
#define CF      2048
#define RANK    32

// ---- CDNA5 async global->LDS copy (ISA 15.18.3, opcode 98, GV mode) --------
__device__ __forceinline__ void async_copy_b128(unsigned lds_addr, const void* gaddr) {
  asm volatile("global_load_async_to_lds_b128 %0, %1, off"
               :: "v"(lds_addr), "v"((unsigned long long)(uintptr_t)gaddr)
               : "memory");
}
__device__ __forceinline__ void async_wait0() {
  asm volatile("s_wait_asynccnt 0x0" ::: "memory");
}

// ---- WMMA fragment loaders (ISA 7.12.2 16-bit layouts, wave32) -------------
// A-style: element (row = lane&15, K); K per VGPR r, half = lane>>4:
//   r in 0..7 : K0 = (r>>2)*16 + half*8 + (r&3)*2
__device__ __forceinline__ v16bf load_frag_rowK(const __bf16* base, int row_stride) {
  int lane = threadIdx.x & 31;
  const __bf16* p = base + (size_t)(lane & 15) * row_stride;
  int h8 = (lane >> 4) << 3;
  v16bf f;
#pragma unroll
  for (int r = 0; r < 8; ++r) {
    int k0 = ((r >> 2) << 4) + h8 + ((r & 3) << 1);
    f[2 * r]     = p[k0];
    f[2 * r + 1] = p[k0 + 1];
  }
  return f;
}

// B-style for a K-major source: element (K, col = lane&15), source row-major [K][ld]
__device__ __forceinline__ v16bf load_frag_colK(const __bf16* base, int ld) {
  int lane = threadIdx.x & 31;
  const __bf16* p = base + (lane & 15);
  int h8 = (lane >> 4) << 3;
  v16bf f;
#pragma unroll
  for (int r = 0; r < 8; ++r) {
    int k0 = ((r >> 2) << 4) + h8 + ((r & 3) << 1);
    f[2 * r]     = p[(size_t)k0 * ld];
    f[2 * r + 1] = p[(size_t)(k0 + 1) * ld];
  }
  return f;
}

// ---------------------------------------------------------------------------
// Generic bf16 GEMM: out[M,N] = A[M,K] @ Wt[N,K]^T  (+bias +lora +residual)
// 256 threads (8 waves), 128x128 macro tile, K-step 32.
// Wave grid 4(M)x2(N): each wave owns 32x64 -> 8 WMMAs per K-step.
// ---------------------------------------------------------------------------
__global__ __launch_bounds__(256) void gemm_bf16_kernel(
    const __bf16* __restrict__ A, const __bf16* __restrict__ Wt,
    const float* __restrict__ bias,
    const float* __restrict__ addend, float addScale,
    const __bf16* __restrict__ loraT, const __bf16* __restrict__ loraB, float loraScale,
    float* __restrict__ outF, __bf16* __restrict__ outB, float outScale,
    int Mr, int Nc, int K)
{
  __shared__ __bf16 As[128][32];   // 8 KB
  __shared__ __bf16 Bs[128][32];   // 8 KB
  int tid  = threadIdx.x;
  int wave = tid >> 5, lane = tid & 31;
  int bm = blockIdx.x * 128;
  int bn = blockIdx.y * 128;
  int wm = (wave & 3) * 32;     // wave's M sub-tile (2 frags)
  int wn = (wave >> 2) * 64;    // wave's N sub-tile (4 frags)
  v8f acc[2][4] = {};

  int ldr = (tid * 16) >> 5;    // staging row (0..127)
  int ldc = (tid * 16) & 31;    // staging col (0 or 16)
  int ar = min(bm + ldr, Mr - 1);
  int br = min(bn + ldr, Nc - 1);
  const __bf16* Aall = A  + (size_t)ar * K + ldc;
  const __bf16* Wall = Wt + (size_t)br * K + ldc;

  for (int k0 = 0; k0 < K; k0 += 32) {
    *(u32x4*)&As[ldr][ldc]     = *(const u32x4*)(Aall + k0);
    *(u32x4*)&As[ldr][ldc + 8] = *(const u32x4*)(Aall + k0 + 8);
    *(u32x4*)&Bs[ldr][ldc]     = *(const u32x4*)(Wall + k0);
    *(u32x4*)&Bs[ldr][ldc + 8] = *(const u32x4*)(Wall + k0 + 8);
    if (k0 + 32 < K) {          // pull next panels toward L2/L0 early
      __builtin_prefetch(Aall + k0 + 32, 0, 1);
      __builtin_prefetch(Wall + k0 + 32, 0, 1);
    }
    __syncthreads();
    v16bf a0 = load_frag_rowK(&As[wm][0], 32);
    v16bf a1 = load_frag_rowK(&As[wm + 16][0], 32);
    v16bf b0 = load_frag_rowK(&Bs[wn][0], 32);
    v16bf b1 = load_frag_rowK(&Bs[wn + 16][0], 32);
    v16bf b2 = load_frag_rowK(&Bs[wn + 32][0], 32);
    v16bf b3 = load_frag_rowK(&Bs[wn + 48][0], 32);
    acc[0][0] = __builtin_amdgcn_wmma_f32_16x16x32_bf16(false, a0, false, b0, (short)0, acc[0][0], false, false);
    acc[0][1] = __builtin_amdgcn_wmma_f32_16x16x32_bf16(false, a0, false, b1, (short)0, acc[0][1], false, false);
    acc[0][2] = __builtin_amdgcn_wmma_f32_16x16x32_bf16(false, a0, false, b2, (short)0, acc[0][2], false, false);
    acc[0][3] = __builtin_amdgcn_wmma_f32_16x16x32_bf16(false, a0, false, b3, (short)0, acc[0][3], false, false);
    acc[1][0] = __builtin_amdgcn_wmma_f32_16x16x32_bf16(false, a1, false, b0, (short)0, acc[1][0], false, false);
    acc[1][1] = __builtin_amdgcn_wmma_f32_16x16x32_bf16(false, a1, false, b1, (short)0, acc[1][1], false, false);
    acc[1][2] = __builtin_amdgcn_wmma_f32_16x16x32_bf16(false, a1, false, b2, (short)0, acc[1][2], false, false);
    acc[1][3] = __builtin_amdgcn_wmma_f32_16x16x32_bf16(false, a1, false, b3, (short)0, acc[1][3], false, false);
    __syncthreads();
  }

  int colL = lane & 15, half = lane >> 4;
#pragma unroll
  for (int i = 0; i < 2; ++i) {
#pragma unroll
    for (int j = 0; j < 4; ++j) {
      int n = bn + wn + j * 16 + colL;
      if (n >= Nc) continue;
#pragma unroll
      for (int r = 0; r < 8; ++r) {
        int m = bm + wm + i * 16 + half * 8 + r;
        if (m >= Mr) continue;
        float v = acc[i][j][r] * outScale;
        if (bias) v += bias[n];
        if (loraT) {  // rank-32 LoRA epilogue: v += s * <loraT[m,:], loraB[n,:]>
          const __bf16* tp = loraT + (size_t)m * RANK;
          const __bf16* lb = loraB + (size_t)n * RANK;
          float s = 0.f;
#pragma unroll
          for (int rr = 0; rr < RANK; ++rr) s += (float)tp[rr] * (float)lb[rr];
          v += loraScale * s;
        }
        size_t idx = (size_t)m * Nc + n;
        if (addend) v += addScale * addend[idx];
        if (outF) outF[idx] = v;
        if (outB) outB[idx] = (__bf16)v;
      }
    }
  }
}

// ---------------------------------------------------------------------------
// Fused flash attention: Q[b,h,N,64] x K/V[b,h,M,64] -> out packed [b,n,C]
// Per block: one (b,h), 128 query rows (16 per wave). K/V chunks of 128 rows
// staged in LDS by GLOBAL_LOAD_ASYNC_TO_LDS (ASYNCcnt); online softmax in
// registers; P transposed via per-wave LDS scratch into WMMA A layout.
// ---------------------------------------------------------------------------
__global__ __launch_bounds__(256) void attn_kernel(
    const __bf16* __restrict__ qp, const __bf16* __restrict__ kp,
    const __bf16* __restrict__ vp, __bf16* __restrict__ outp)
{
  __shared__ __bf16 Ks[128][64];
  __shared__ __bf16 Vs[128][64];
  __shared__ __bf16 Ps[8][16][32];

  int bh   = blockIdx.x;                 // b*HEADS + h
  int tid  = threadIdx.x;
  int wave = tid >> 5, lane = tid & 31;
  int colL = lane & 15, half = lane >> 4;
  int qrow0 = blockIdx.y * 128 + wave * 16;

  const __bf16* Qb = qp + ((size_t)bh * N_TOK + qrow0) * HD;
  const __bf16* Kb = kp + (size_t)bh * M_TOK * HD;
  const __bf16* Vb = vp + (size_t)bh * M_TOK * HD;

  v16bf qa0 = load_frag_rowK(Qb, HD);        // channels 0..31
  v16bf qa1 = load_frag_rowK(Qb + 32, HD);   // channels 32..63

  float mrow[8], lrow[8];
  v8f oacc[4] = {};
#pragma unroll
  for (int r = 0; r < 8; ++r) { mrow[r] = -1e30f; lrow[r] = 0.f; }

  int sr = (tid * 8) >> 6;        // this thread's staging row (0..31) base
  int sc = (tid * 8) & 63;        // staging col

  for (int m0 = 0; m0 < M_TOK; m0 += 128) {
#pragma unroll
    for (int i = 0; i < 4; ++i) {           // async-stage 128x64 bf16 K and V
      int r = sr + i * 32;
      async_copy_b128((unsigned)(uintptr_t)&Ks[r][sc], Kb + (size_t)(m0 + r) * HD + sc);
      async_copy_b128((unsigned)(uintptr_t)&Vs[r][sc], Vb + (size_t)(m0 + r) * HD + sc);
    }
    async_wait0();
    __syncthreads();

    for (int mc = 0; mc < 128; mc += 32) {
      // S = Q.K^T for 32 keys (two 16x16 tiles), 4 WMMAs
      v8f s0 = {}, s1 = {};
      s0 = __builtin_amdgcn_wmma_f32_16x16x32_bf16(false, qa0, false, load_frag_rowK(&Ks[mc][0], HD),       (short)0, s0, false, false);
      s0 = __builtin_amdgcn_wmma_f32_16x16x32_bf16(false, qa1, false, load_frag_rowK(&Ks[mc][32], HD),      (short)0, s0, false, false);
      s1 = __builtin_amdgcn_wmma_f32_16x16x32_bf16(false, qa0, false, load_frag_rowK(&Ks[mc + 16][0], HD),  (short)0, s1, false, false);
      s1 = __builtin_amdgcn_wmma_f32_16x16x32_bf16(false, qa1, false, load_frag_rowK(&Ks[mc + 16][32], HD), (short)0, s1, false, false);

      // online softmax over these 32 keys (row = half*8 + r, col = lane&15)
      float corr[8];
#pragma unroll
      for (int r = 0; r < 8; ++r) {
        float a0 = s0[r] * 0.125f, a1 = s1[r] * 0.125f;  // scale = hd^-0.5
        float mx = fmaxf(a0, a1);
#pragma unroll
        for (int off = 8; off; off >>= 1) mx = fmaxf(mx, __shfl_xor(mx, off, 32));
        float mnew = fmaxf(mrow[r], mx);
        float cc = __expf(mrow[r] - mnew);
        float p0 = __expf(a0 - mnew), p1 = __expf(a1 - mnew);
        float rs = p0 + p1;
#pragma unroll
        for (int off = 8; off; off >>= 1) rs += __shfl_xor(rs, off, 32);
        lrow[r] = lrow[r] * cc + rs;
        mrow[r] = mnew;
        corr[r] = cc;
        Ps[wave][half * 8 + r][colL]      = (__bf16)p0;   // transpose P to A layout
        Ps[wave][half * 8 + r][16 + colL] = (__bf16)p1;
      }
#pragma unroll
      for (int t = 0; t < 4; ++t)
#pragma unroll
        for (int r = 0; r < 8; ++r) oacc[t][r] *= corr[r];

      v16bf pa = load_frag_rowK(&Ps[wave][0][0], 32);     // 16x32 P fragment
#pragma unroll
      for (int t = 0; t < 4; ++t)                         // O += P.V (4 d-tiles)
        oacc[t] = __builtin_amdgcn_wmma_f32_16x16x32_bf16(
            false, pa, false, load_frag_colK(&Vs[mc][t * 16], HD), (short)0, oacc[t], false, false);
    }
    __syncthreads();
  }

  int b = bh >> 3, hh = bh & 7;
  float inv[8];
#pragma unroll
  for (int r = 0; r < 8; ++r) inv[r] = 1.0f / lrow[r];
#pragma unroll
  for (int t = 0; t < 4; ++t)
#pragma unroll
    for (int r = 0; r < 8; ++r) {
      int row = qrow0 + half * 8 + r;
      int d   = hh * HD + t * 16 + colL;
      outp[((size_t)b * N_TOK + row) * C_DIM + d] = (__bf16)(oacc[t][r] * inv[r]);
    }
}

// ---------------------------------------------------------------------------
// Elementwise helper kernels
// ---------------------------------------------------------------------------
__global__ void cvt_kernel(const float* __restrict__ in, __bf16* __restrict__ out, int n) {
  int i = blockIdx.x * 256 + threadIdx.x;
  if (i < n) out[i] = (__bf16)in[i];
}

// LayerNorm over C=512, one row per 256-thread block, bf16 output
__global__ __launch_bounds__(256) void ln_kernel(
    const float* __restrict__ x, const float* __restrict__ g,
    const float* __restrict__ b, __bf16* __restrict__ out)
{
  __shared__ float s1[256], s2[256];
  int row = blockIdx.x, tid = threadIdx.x;
  const float* xr = x + (size_t)row * C_DIM;
  float v0 = xr[tid], v1 = xr[tid + 256];
  s1[tid] = v0 + v1;
  s2[tid] = v0 * v0 + v1 * v1;
  __syncthreads();
  for (int s = 128; s > 0; s >>= 1) {
    if (tid < s) { s1[tid] += s1[tid + s]; s2[tid] += s2[tid + s]; }
    __syncthreads();
  }
  float mean = s1[0] * (1.0f / C_DIM);
  float var  = s2[0] * (1.0f / C_DIM) - mean * mean;
  float inv  = rsqrtf(var + 1e-5f);
  __bf16* o = out + (size_t)row * C_DIM;
  o[tid]       = (__bf16)((v0 - mean) * inv * g[tid]       + b[tid]);
  o[tid + 256] = (__bf16)((v1 - mean) * inv * g[tid + 256] + b[tid + 256]);
}

// im2col for 2x2/stride-2 conv: xp[b*M+m][c*4+ky*2+kx] = h1[b, (2i+ky)*W+(2j+kx), c]
__global__ void im2col_kernel(const __bf16* __restrict__ h1, __bf16* __restrict__ xp) {
  size_t idx = (size_t)blockIdx.x * 256 + threadIdx.x;
  if (idx >= (size_t)B_SZ * M_TOK * 2048) return;
  int e = idx & 2047;
  size_t bm = idx >> 11;
  int c = e >> 2, ky = (e >> 1) & 1, kx = e & 1;
  int m = bm & (M_TOK - 1), b = bm >> 10;
  int i = m >> 5, j = m & 31;                  // W/SR = 32
  int n = (2 * i + ky) * WID + (2 * j + kx);
  xp[idx] = h1[((size_t)b * N_TOK + n) * C_DIM + c];
}

// pack q f32 [b,n,C] -> bf16 [b,h,n,d]
__global__ void pack_q_kernel(const float* __restrict__ q, __bf16* __restrict__ qp) {
  size_t idx = (size_t)blockIdx.x * 256 + threadIdx.x;
  if (idx >= (size_t)B_SZ * HEADS * N_TOK * HD) return;
  int d = idx & 63; size_t t = idx >> 6;
  int n = t & (N_TOK - 1); t >>= 12;
  int h = t & 7; int b = t >> 3;
  qp[idx] = (__bf16)q[((size_t)b * N_TOK + n) * C_DIM + h * HD + d];
}

// split kv f32 [b,m,2C] into k/v bf16 [b,h,m,d]; v gets LoRA (raw reshape)
__global__ void pack_kv_kernel(const float* __restrict__ kv, const float* __restrict__ lv,
                               __bf16* __restrict__ kb, __bf16* __restrict__ vb) {
  size_t idx = (size_t)blockIdx.x * 256 + threadIdx.x;
  if (idx >= (size_t)B_SZ * HEADS * M_TOK * HD) return;
  int d = idx & 63; size_t t = idx >> 6;
  int m = t & (M_TOK - 1); t >>= 10;
  int h = t & 7; int b = t >> 3;
  size_t kvrow = ((size_t)b * M_TOK + m) * (2 * C_DIM);
  kb[idx] = (__bf16)kv[kvrow + h * HD + d];
  float vv = kv[kvrow + C_DIM + h * HD + d];
  // torch reshape [B,M,C]->[B,HEAD,M,d] without permute:
  vv += lv[(size_t)b * M_TOK * C_DIM + (size_t)h * (M_TOK * HD) + (size_t)m * HD + d];
  vb[idx] = (__bf16)vv;
}

// depthwise 3x3 (pad 1) on f [b,n,Cf] channel-last + exact GELU -> bf16
__global__ void dwconv_gelu_kernel(const float* __restrict__ F, const float* __restrict__ dw_w,
                                   const float* __restrict__ dw_b, __bf16* __restrict__ G) {
  size_t idx = (size_t)blockIdx.x * 256 + threadIdx.x;
  if (idx >= (size_t)B_SZ * N_TOK * CF) return;
  int c = idx & (CF - 1); size_t t = idx >> 11;
  int n = t & (N_TOK - 1); int b = t >> 12;
  int y = n >> 6, x = n & 63;
  float acc = dw_b[c];
#pragma unroll
  for (int ky = 0; ky < 3; ++ky)
#pragma unroll
    for (int kx = 0; kx < 3; ++kx) {
      int yy = y + ky - 1, xx = x + kx - 1;
      if (yy >= 0 && yy < HGT && xx >= 0 && xx < WID)
        acc += F[((size_t)b * N_TOK + yy * WID + xx) * CF + c] * dw_w[c * 9 + ky * 3 + kx];
    }
  float ge = 0.5f * acc * (1.0f + erff(acc * 0.70710678118654752f));
  G[idx] = (__bf16)ge;
}

// ---------------------------------------------------------------------------
extern "C" void kernel_launch(void* const* d_in, const int* in_sizes, int n_in,
                              void* d_out, int out_size, void* d_ws, size_t ws_size,
                              hipStream_t stream) {
  (void)in_sizes; (void)n_in; (void)out_size; (void)ws_size;
  const float* x       = (const float*)d_in[0];
  const float* norm1_w = (const float*)d_in[3];
  const float* norm1_b = (const float*)d_in[4];
  const float* q_w     = (const float*)d_in[5];
  const float* q_b     = (const float*)d_in[6];
  const float* kv_w    = (const float*)d_in[7];
  const float* kv_b    = (const float*)d_in[8];
  const float* proj_w  = (const float*)d_in[9];
  const float* proj_b  = (const float*)d_in[10];
  const float* lqA     = (const float*)d_in[11];
  const float* lqB     = (const float*)d_in[12];
  const float* lvA     = (const float*)d_in[13];
  const float* lvB     = (const float*)d_in[14];
  const float* sr_w    = (const float*)d_in[15];
  const float* sr_b    = (const float*)d_in[16];
  const float* srn_w   = (const float*)d_in[17];
  const float* srn_b   = (const float*)d_in[18];
  const float* norm2_w = (const float*)d_in[19];
  const float* norm2_b = (const float*)d_in[20];
  const float* fc1_w   = (const float*)d_in[21];
  const float* fc1_b   = (const float*)d_in[22];
  const float* lf1A    = (const float*)d_in[23];
  const float* lf1B    = (const float*)d_in[24];
  const float* dw_w    = (const float*)d_in[25];
  const float* dw_b    = (const float*)d_in[26];
  const float* fc2_w   = (const float*)d_in[27];
  const float* fc2_b   = (const float*)d_in[28];
  const float* lf2A    = (const float*)d_in[29];
  const float* lf2B    = (const float*)d_in[30];

  const size_t BN = (size_t)B_SZ * N_TOK;   // 8192
  const size_t BM = (size_t)B_SZ * M_TOK;   // 2048

  char* ws = (char*)d_ws;
  size_t off = 0;
  auto alloc = [&](size_t bytes) { char* p = ws + off; off += (bytes + 255) & ~(size_t)255; return (void*)p; };

  // bf16 weight copies
  __bf16* qw_bf   = (__bf16*)alloc(C_DIM * C_DIM * 2);
  __bf16* kvw_bf  = (__bf16*)alloc(2 * C_DIM * C_DIM * 2);
  __bf16* prw_bf  = (__bf16*)alloc(C_DIM * C_DIM * 2);
  __bf16* srw_bf  = (__bf16*)alloc(C_DIM * 2048 * 2);
  __bf16* f1w_bf  = (__bf16*)alloc((size_t)CF * C_DIM * 2);
  __bf16* f2w_bf  = (__bf16*)alloc((size_t)C_DIM * CF * 2);
  __bf16* lqA_bf  = (__bf16*)alloc(RANK * C_DIM * 2);
  __bf16* lqB_bf  = (__bf16*)alloc(C_DIM * RANK * 2);
  __bf16* lvA_bf  = (__bf16*)alloc(RANK * C_DIM * 2);
  __bf16* lvB_bf  = (__bf16*)alloc(C_DIM * RANK * 2);
  __bf16* l1A_bf  = (__bf16*)alloc(RANK * C_DIM * 2);
  __bf16* l1B_bf  = (__bf16*)alloc((size_t)CF * RANK * 2);
  __bf16* l2A_bf  = (__bf16*)alloc((size_t)RANK * CF * 2);
  __bf16* l2B_bf  = (__bf16*)alloc(C_DIM * RANK * 2);
  // activations
  __bf16* h1   = (__bf16*)alloc(BN * C_DIM * 2);
  __bf16* tq   = (__bf16*)alloc(BN * RANK * 2);
  float*  qf   = (float*)alloc(BN * C_DIM * 4);
  __bf16* qp   = (__bf16*)alloc(BN * C_DIM * 2);
  __bf16* xpat = (__bf16*)alloc(BM * 2048 * 2);
  float*  xs   = (float*)alloc(BM * C_DIM * 4);
  __bf16* xsln = (__bf16*)alloc(BM * C_DIM * 2);
  __bf16* tv   = (__bf16*)alloc(BM * RANK * 2);
  float*  lv   = (float*)alloc(BM * C_DIM * 4);
  float*  kvo  = (float*)alloc(BM * 2 * C_DIM * 4);
  __bf16* kbf  = (__bf16*)alloc(BM * C_DIM * 2);
  __bf16* vbf  = (__bf16*)alloc(BM * C_DIM * 2);
  __bf16* ao   = (__bf16*)alloc(BN * C_DIM * 2);
  float*  x1   = (float*)alloc(BN * C_DIM * 4);
  __bf16* h2   = (__bf16*)alloc(BN * C_DIM * 2);
  __bf16* t1   = (__bf16*)alloc(BN * RANK * 2);
  float*  Fbuf = (float*)alloc(BN * CF * 4);
  __bf16* Gbuf = (__bf16*)alloc(BN * CF * 2);
  __bf16* t2   = (__bf16*)alloc(BN * RANK * 2);

  auto cvt = [&](const float* src, __bf16* dst, size_t n) {
    cvt_kernel<<<(unsigned)((n + 255) / 256), 256, 0, stream>>>(src, dst, (int)n);
  };
  auto gemm = [&](const __bf16* A, const __bf16* Wt, const float* bias,
                  const float* addend, float addScale,
                  const __bf16* loraT, const __bf16* loraB, float loraScale,
                  float* outF, __bf16* outB, float outScale, int Mr, int Nc, int K) {
    dim3 g(Mr / 128, (Nc + 127) / 128);
    gemm_bf16_kernel<<<g, 256, 0, stream>>>(A, Wt, bias, addend, addScale,
                                            loraT, loraB, loraScale, outF, outB, outScale, Mr, Nc, K);
  };

  // weight conversion
  cvt(q_w, qw_bf, (size_t)C_DIM * C_DIM);
  cvt(kv_w, kvw_bf, (size_t)2 * C_DIM * C_DIM);
  cvt(proj_w, prw_bf, (size_t)C_DIM * C_DIM);
  cvt(sr_w, srw_bf, (size_t)C_DIM * 2048);
  cvt(fc1_w, f1w_bf, (size_t)CF * C_DIM);
  cvt(fc2_w, f2w_bf, (size_t)C_DIM * CF);
  cvt(lqA, lqA_bf, RANK * C_DIM); cvt(lqB, lqB_bf, C_DIM * RANK);
  cvt(lvA, lvA_bf, RANK * C_DIM); cvt(lvB, lvB_bf, C_DIM * RANK);
  cvt(lf1A, l1A_bf, RANK * C_DIM); cvt(lf1B, l1B_bf, (size_t)CF * RANK);
  cvt(lf2A, l2A_bf, (size_t)RANK * CF); cvt(lf2B, l2B_bf, C_DIM * RANK);

  // ---- attention branch ----
  ln_kernel<<<(unsigned)BN, 256, 0, stream>>>(x, norm1_w, norm1_b, h1);
  gemm(h1, lqA_bf, nullptr, nullptr, 0.f, nullptr, nullptr, 0.f,
       nullptr, tq, 1.f, (int)BN, RANK, C_DIM);
  gemm(h1, qw_bf, q_b, nullptr, 0.f, tq, lqB_bf, 0.125f,
       qf, nullptr, 1.f, (int)BN, C_DIM, C_DIM);
  pack_q_kernel<<<(unsigned)(BN * C_DIM / 256), 256, 0, stream>>>(qf, qp);

  im2col_kernel<<<(unsigned)(BM * 2048 / 256), 256, 0, stream>>>(h1, xpat);
  gemm(xpat, srw_bf, sr_b, nullptr, 0.f, nullptr, nullptr, 0.f,
       xs, nullptr, 1.f, (int)BM, C_DIM, 2048);
  ln_kernel<<<(unsigned)BM, 256, 0, stream>>>(xs, srn_w, srn_b, xsln);

  gemm(xsln, lvA_bf, nullptr, nullptr, 0.f, nullptr, nullptr, 0.f,
       nullptr, tv, 1.f, (int)BM, RANK, C_DIM);
  gemm(tv, lvB_bf, nullptr, nullptr, 0.f, nullptr, nullptr, 0.f,
       lv, nullptr, 0.125f, (int)BM, C_DIM, RANK);
  gemm(xsln, kvw_bf, kv_b, nullptr, 0.f, nullptr, nullptr, 0.f,
       kvo, nullptr, 1.f, (int)BM, 2 * C_DIM, C_DIM);
  pack_kv_kernel<<<(unsigned)(BM * C_DIM / 256), 256, 0, stream>>>(kvo, lv, kbf, vbf);

  attn_kernel<<<dim3(B_SZ * HEADS, N_TOK / 128), 256, 0, stream>>>(qp, kbf, vbf, ao);

  gemm(ao, prw_bf, proj_b, x, 1.f, nullptr, nullptr, 0.f,
       x1, nullptr, 1.f, (int)BN, C_DIM, C_DIM);

  // ---- MLP branch ----
  ln_kernel<<<(unsigned)BN, 256, 0, stream>>>(x1, norm2_w, norm2_b, h2);
  gemm(h2, l1A_bf, nullptr, nullptr, 0.f, nullptr, nullptr, 0.f,
       nullptr, t1, 1.f, (int)BN, RANK, C_DIM);
  gemm(h2, f1w_bf, fc1_b, nullptr, 0.f, t1, l1B_bf, 0.125f,
       Fbuf, nullptr, 1.f, (int)BN, CF, C_DIM);
  dwconv_gelu_kernel<<<(unsigned)(BN * CF / 256), 256, 0, stream>>>(Fbuf, dw_w, dw_b, Gbuf);
  gemm(Gbuf, l2A_bf, nullptr, nullptr, 0.f, nullptr, nullptr, 0.f,
       nullptr, t2, 1.f, (int)BN, RANK, CF);
  gemm(Gbuf, f2w_bf, fc2_b, x1, 1.f, t2, l2B_bf, 0.125f,
       (float*)d_out, nullptr, 1.f, (int)BN, C_DIM, CF);
}
